// ChunkAttentionWithRelativeRightContext_31525059953360
// MI455X (gfx1250) — matrix operand order repaired and checked by reading, not compiled, not run on hardware
//
#include <hip/hip_runtime.h>

// ---------------------------------------------------------------------------
// Transformer-XL relative attention for MI455X (gfx1250, wave32, WMMA).
// bf16 WMMA (v_wmma_f32_16x16x32_bf16) for all GEMMs, fp32 accumulation.
// Attention is fully fused flash-style: scores (+rel_shift via 2 extra WMMAs
// and ds_bpermute diagonal extraction), online softmax (ds_bpermute butterfly
// row reductions), and P@V (LDS-transposed P fragments) in one kernel.
// ---------------------------------------------------------------------------

typedef long long i64;

static constexpr int H = 8, DK = 64, FEAT = 512;
static constexpr int Bsz = 8, T1 = 512, CACHE_T = 1024, T2 = T1 + CACHE_T; // 1536
static constexpr int NPOS = T1 + T2 - 1;  // 2047
static constexpr int NPAD = 2048;         // padded position count
static constexpr int BH = Bsz * H;        // 64

typedef __bf16 bf16x16 __attribute__((ext_vector_type(16)));
typedef float  f32x8   __attribute__((ext_vector_type(8)));

union FragU { bf16x16 f; uint4 q[2]; };

#define DEV static __device__ __forceinline__
#define NEG_INF (-__builtin_inff())

DEV unsigned short bf16b(float x) {
  unsigned u = __float_as_uint(x);
  u += 0x7fffu + ((u >> 16) & 1u);        // round-to-nearest-even
  return (unsigned short)(u >> 16);
}

// A fragment: 16x32 bf16, row = m0 + (lane&15);
// lanes 0-15 hold K = k0+{0..7, 16..23}, lanes 16-31 hold K = k0+{8..15, 24..31}
DEV bf16x16 load_a(const unsigned short* A, i64 lda, i64 m0, int k0, int lane) {
  i64 row = m0 + (lane & 15);
  const unsigned short* p = A + row * lda + k0 + ((lane >= 16) ? 8 : 0);
  FragU u;
  u.q[0] = *(const uint4*)p;
  u.q[1] = *(const uint4*)(p + 16);
  return u.f;
}

// B fragment for C = A @ Bsrc^T where Bsrc is (N x K) row-major:
// lane column n = n0 + (lane&15); lanes 0-15 hold K = k0+{0..7, 16..23}, lanes 16-31 hold K = k0+{8..15, 24..31}.
DEV bf16x16 load_bt(const unsigned short* Bm, i64 ldb, i64 n0, int k0, int lane) {
  i64 row = n0 + (lane & 15);
  const unsigned short* p = Bm + row * ldb + k0 + ((lane >= 16) ? 8 : 0);
  FragU u;
  u.q[0] = *(const uint4*)p;
  u.q[1] = *(const uint4*)(p + 16);
  return u.f;
}

DEV f32x8 wmma_bf16(bf16x16 a, bf16x16 b, f32x8 c) {
  return __builtin_amdgcn_wmma_f32_16x16x32_bf16(false, a, false, b,
                                                 (short)0, c, false, false);
}

// One wave computes a 16x64 output strip (A frag reused across 4 B tiles).
DEV void mm_row4(const unsigned short* A, i64 lda, const unsigned short* Bt, i64 ldb,
                 i64 m0, i64 n0, int K, int lane, f32x8 acc[4]) {
  for (int k0 = 0; k0 < K; k0 += 32) {
    bf16x16 a = load_a(A, lda, m0, k0, lane);
#pragma unroll
    for (int t = 0; t < 4; t++) {
      bf16x16 b = load_bt(Bt, ldb, n0 + 16 * t, k0, lane);
      acc[t] = wmma_bf16(a, b, acc[t]);
    }
  }
}

DEV float lane_gather(float v, int srcLane) {
  return __int_as_float(__builtin_amdgcn_ds_bpermute(srcLane << 2, __float_as_int(v)));
}

// --------------------------- conversion kernels ---------------------------

__global__ void k_cvt(const float* __restrict__ s, unsigned short* __restrict__ d, int n) {
  int i = blockIdx.x * blockDim.x + threadIdx.x;
  if (i < n) d[i] = bf16b(s[i]);
}

__global__ void k_cvt_pad(const float* __restrict__ s, unsigned short* __restrict__ d,
                          int n_src, int n_tot) {
  int i = blockIdx.x * blockDim.x + threadIdx.x;
  if (i < n_tot) d[i] = (i < n_src) ? bf16b(s[i]) : (unsigned short)0;
}

// Copy cache into new_cache[:, :, :CACHE_T, :] and build bf16 k / v^T copies.
__global__ void k_cache(const float* __restrict__ cache, float* __restrict__ new_cache,
                        unsigned short* __restrict__ kb, unsigned short* __restrict__ vT) {
  i64 i = (i64)blockIdx.x * blockDim.x + threadIdx.x;
  if (i >= (i64)BH * CACHE_T * 2 * DK) return;
  int c = (int)(i & 127);
  i64 rest = i >> 7;
  int t = (int)(rest & (CACHE_T - 1));
  i64 bh = rest >> 10;
  float val = cache[i];
  new_cache[(bh * T2 + t) * (2 * DK) + c] = val;
  if (c < DK) kb[(bh * T2 + t) * DK + c] = bf16b(val);
  else        vT[(bh * DK + (c - DK)) * T2 + t] = bf16b(val);
}

// ------------------------------ projections -------------------------------

__global__ void k_qproj(const unsigned short* __restrict__ Abf,
                        const unsigned short* __restrict__ Wbf,
                        const float* __restrict__ bias,
                        const float* __restrict__ pbu, const float* __restrict__ pbv,
                        unsigned short* __restrict__ qu, unsigned short* __restrict__ qv) {
  int lane = threadIdx.x & 31;
  int gw = blockIdx.x * (blockDim.x >> 5) + (threadIdx.x >> 5);
  i64 m0 = (i64)(gw >> 3) * 16;
  i64 n0 = (i64)(gw & 7) * 64;
  f32x8 acc[4] = {};
  mm_row4(Abf, FEAT, Wbf, FEAT, m0, n0, FEAT, lane, acc);
  int col = lane & 15;
#pragma unroll
  for (int t = 0; t < 4; t++) {
    i64 n = n0 + t * 16 + col;
    int h = (int)(n >> 6), dk = (int)(n & 63);
    float bq = bias[n], bu = pbu[n], bv = pbv[n];
#pragma unroll
    for (int r = 0; r < 8; r++) {
      i64 m = m0 + ((lane < 16) ? r : r + 8);
      int b = (int)(m >> 9), tt = (int)(m & (T1 - 1));
      float c = acc[t][r] + bq;
      i64 o = (((i64)(b * H + h) * T1) + tt) * DK + dk;
      qu[o] = bf16b(c + bu);
      qv[o] = bf16b(c + bv);
    }
  }
}

__global__ void k_kvproj(const unsigned short* __restrict__ Abf,
                         const unsigned short* __restrict__ Wbf,
                         const float* __restrict__ bias,
                         float* __restrict__ new_cache,
                         unsigned short* __restrict__ kb,
                         unsigned short* __restrict__ vT, int is_v) {
  int lane = threadIdx.x & 31;
  int gw = blockIdx.x * (blockDim.x >> 5) + (threadIdx.x >> 5);
  i64 m0 = (i64)(gw >> 3) * 16;
  i64 n0 = (i64)(gw & 7) * 64;
  f32x8 acc[4] = {};
  mm_row4(Abf, FEAT, Wbf, FEAT, m0, n0, FEAT, lane, acc);
  int col = lane & 15;
#pragma unroll
  for (int t = 0; t < 4; t++) {
    i64 n = n0 + t * 16 + col;
    int h = (int)(n >> 6), dk = (int)(n & 63);
    float bs = bias[n];
#pragma unroll
    for (int r = 0; r < 8; r++) {
      i64 m = m0 + ((lane < 16) ? r : r + 8);
      int b = (int)(m >> 9), tt = (int)(m & (T1 - 1));
      float c = acc[t][r] + bs;
      i64 bh = b * H + h;
      i64 tpos = CACHE_T + tt;
      new_cache[(bh * T2 + tpos) * (2 * DK) + (i64)is_v * DK + dk] = c;
      if (!is_v) kb[(bh * T2 + tpos) * DK + dk] = bf16b(c);
      else       vT[(bh * DK + dk) * T2 + tpos] = bf16b(c);
    }
  }
}

__global__ void k_pproj(const unsigned short* __restrict__ Abf,
                        const unsigned short* __restrict__ Wbf,
                        unsigned short* __restrict__ pdst) {
  int lane = threadIdx.x & 31;
  int gw = blockIdx.x * (blockDim.x >> 5) + (threadIdx.x >> 5);
  i64 m0 = (i64)(gw >> 3) * 16;   // 128 m-tiles over NPAD
  i64 n0 = (i64)(gw & 7) * 64;
  f32x8 acc[4] = {};
  mm_row4(Abf, FEAT, Wbf, FEAT, m0, n0, FEAT, lane, acc);
  int col = lane & 15;
#pragma unroll
  for (int t = 0; t < 4; t++) {
    i64 n = n0 + t * 16 + col;
    int h = (int)(n >> 6), dk = (int)(n & 63);
#pragma unroll
    for (int r = 0; r < 8; r++) {
      i64 m = m0 + ((lane < 16) ? r : r + 8);
      pdst[((i64)h * NPAD + m) * DK + dk] = bf16b(acc[t][r]);
    }
  }
}

// ---------------- fused attention: scores + softmax + P@V ------------------
// One wave owns a 16-row query tile; O (16x64 fp32) and per-row (m,l) stay in
// registers across all 96 key tiles. j-tiles processed in pairs so P is a
// full 16x32 bf16 A-fragment (K=32) for the P@V WMMAs.

__global__ void k_attn(const unsigned short* __restrict__ qu,
                       const unsigned short* __restrict__ qv,
                       const unsigned short* __restrict__ kb,
                       const unsigned short* __restrict__ pb,
                       const unsigned short* __restrict__ vT,
                       const unsigned char* __restrict__ mask,
                       unsigned short* __restrict__ xbf) {
  __shared__ unsigned short pls[4][16 * 32];   // per-wave P staging (1KB each)
  int lane = threadIdx.x & 31;
  int w = threadIdx.x >> 5;
  int mt = blockIdx.x * 4 + w;    // 32 i-tiles over T1
  int bh = blockIdx.z;
  int b = bh >> 3, h = bh & 7;
  i64 m0 = (i64)mt * 16;

  const unsigned short* quB = qu + (i64)bh * T1 * DK;
  const unsigned short* qvB = qv + (i64)bh * T1 * DK;
  const unsigned short* kB  = kb + (i64)bh * T2 * DK;
  const unsigned short* pB  = pb + (i64)h * NPAD * DK;
  const unsigned short* vB  = vT + (i64)bh * DK * T2;

  // Query fragments loaded once (K = 64 -> two 16x32 fragments each).
  bf16x16 aq0 = load_a(quB, DK, m0, 0, lane);
  bf16x16 aq1 = load_a(quB, DK, m0, 32, lane);
  bf16x16 avv0 = load_a(qvB, DK, m0, 0, lane);
  bf16x16 avv1 = load_a(qvB, DK, m0, 32, lane);

  f32x8 O[4] = {};
  float mrow[8], lrow[8];
#pragma unroll
  for (int r = 0; r < 8; r++) { mrow[r] = NEG_INF; lrow[r] = 0.f; }

  int dj = lane & 15;
  int halfbase = lane & 16;
  unsigned short* myP = &pls[w][0];

  for (int jp = 0; jp < T2 / 32; jp++) {       // 48 pairs of j-tiles
    float s[2][8];
#pragma unroll
    for (int c = 0; c < 2; c++) {
      i64 j0 = (i64)(jp * 2 + c) * 16;
      i64 nR = (i64)T1 - 1 - m0 + j0;          // diagonal p-row base
      i64 nL = nR - 15;
      f32x8 ac = {}, aL = {}, aR = {};
      ac = wmma_bf16(aq0,  load_bt(kB, DK, j0, 0,  lane), ac);
      aL = wmma_bf16(avv0, load_bt(pB, DK, nL, 0,  lane), aL);
      aR = wmma_bf16(avv0, load_bt(pB, DK, nR, 0,  lane), aR);
      ac = wmma_bf16(aq1,  load_bt(kB, DK, j0, 32, lane), ac);
      aL = wmma_bf16(avv1, load_bt(pB, DK, nL, 32, lane), aL);
      aR = wmma_bf16(avv1, load_bt(pB, DK, nR, 32, lane), aR);
      unsigned char mb = mask[(i64)b * T2 + j0 + dj];
#pragma unroll
      for (int r = 0; r < 8; r++) {
        int di = r + ((lane >= 16) ? 8 : 0);
        int delta = dj - di;                   // [-15, 15]
        float vR = lane_gather(aR[r], halfbase + (delta & 15));
        float vL = lane_gather(aL[r], halfbase + ((delta + 15) & 15));
        float bd = (delta >= 0) ? vR : vL;
        float sv = (ac[r] + bd) * 0.125f;      // 1/sqrt(DK)
        s[c][r] = mb ? sv : NEG_INF;
      }
    }
    // Online softmax update over these 32 columns (row = half's 16 lanes).
#pragma unroll
    for (int r = 0; r < 8; r++) {
      float tmax = fmaxf(s[0][r], s[1][r]);
#pragma unroll
      for (int d = 1; d < 16; d <<= 1)
        tmax = fmaxf(tmax, lane_gather(tmax, lane ^ d));   // stays in half
      float mNew = fmaxf(mrow[r], tmax);
      bool dead = (mNew == NEG_INF);           // fully-masked so far
      float scale = dead ? 1.f : __expf(mrow[r] - mNew);   // -inf -> 0
      float p0 = dead ? 0.f : __expf(s[0][r] - mNew);
      float p1 = dead ? 0.f : __expf(s[1][r] - mNew);
      float psum = p0 + p1;
#pragma unroll
      for (int d = 1; d < 16; d <<= 1)
        psum += lane_gather(psum, lane ^ d);
      lrow[r] = lrow[r] * scale + psum;
      mrow[r] = mNew;
#pragma unroll
      for (int t = 0; t < 4; t++) O[t][r] *= scale;
      // scatter P into LDS (C layout -> row-major 16x32)
      int row = (lane < 16) ? r : r + 8;
      myP[row * 32 + dj]      = bf16b(p0);
      myP[row * 32 + 16 + dj] = bf16b(p1);
    }
    // Re-read P as an A fragment (16x32, K = the 32 columns of this pair).
    bf16x16 pf;
    {
      int row = lane & 15;
      const unsigned short* pp = myP + row * 32 + ((lane >= 16) ? 8 : 0);
      FragU u;
      u.q[0] = *(const uint4*)pp;
      u.q[1] = *(const uint4*)(pp + 16);
      pf = u.f;
    }
    int tp0 = jp * 32;                         // time offset of this pair
#pragma unroll
    for (int t = 0; t < 4; t++)
      O[t] = wmma_bf16(pf, load_bt(vB, T2, t * 16, tp0, lane), O[t]);
  }

  // Normalize and store x = attn@v in (B, T1, H*DK) bf16 layout.
#pragma unroll
  for (int r = 0; r < 8; r++) {
    float inv = lrow[r] > 0.f ? 1.f / lrow[r] : 0.f;
    i64 m = m0 + ((lane < 16) ? r : r + 8);
#pragma unroll
    for (int t = 0; t < 4; t++) {
      int dk = t * 16 + dj;
      xbf[((i64)(b * T1) + m) * FEAT + h * DK + dk] = bf16b(O[t][r] * inv);
    }
  }
}

// ------------------------------ out = x@Wo^T + bo --------------------------

__global__ void k_out(const unsigned short* __restrict__ xbf,
                      const unsigned short* __restrict__ Wbf,
                      const float* __restrict__ bias, float* __restrict__ out) {
  int lane = threadIdx.x & 31;
  int gw = blockIdx.x * (blockDim.x >> 5) + (threadIdx.x >> 5);
  i64 m0 = (i64)(gw >> 3) * 16;
  i64 n0 = (i64)(gw & 7) * 64;
  f32x8 acc[4] = {};
  mm_row4(xbf, FEAT, Wbf, FEAT, m0, n0, FEAT, lane, acc);
  int col = lane & 15;
#pragma unroll
  for (int t = 0; t < 4; t++) {
    i64 n = n0 + t * 16 + col;
    float bs = bias[n];
#pragma unroll
    for (int r = 0; r < 8; r++) {
      i64 m = m0 + ((lane < 16) ? r : r + 8);
      out[m * FEAT + n] = acc[t][r] + bs;
    }
  }
}

// ------------------------------- launcher ----------------------------------

extern "C" void kernel_launch(void* const* d_in, const int* in_sizes, int n_in,
                              void* d_out, int out_size, void* d_ws, size_t ws_size,
                              hipStream_t stream) {
  const float* query  = (const float*)d_in[0];
  const float* key    = (const float*)d_in[1];
  const float* value  = (const float*)d_in[2];
  const unsigned char* mask = (const unsigned char*)d_in[3];
  const float* pos_emb = (const float*)d_in[4];
  const float* cache   = (const float*)d_in[5];
  const float* Wq = (const float*)d_in[6];  const float* bq = (const float*)d_in[7];
  const float* Wk = (const float*)d_in[8];  const float* bk = (const float*)d_in[9];
  const float* Wv = (const float*)d_in[10]; const float* bv = (const float*)d_in[11];
  const float* Wo = (const float*)d_in[12]; const float* bo = (const float*)d_in[13];
  const float* Wpos = (const float*)d_in[14];
  const float* pbu  = (const float*)d_in[15];
  const float* pbv  = (const float*)d_in[16];

  float* out       = (float*)d_out;
  float* new_cache = out + (i64)Bsz * T1 * FEAT;

  char* ws = (char*)d_ws;
  size_t off = 0;
  auto alloc = [&](size_t elems, size_t esz) -> void* {
    void* p = ws + off;
    off += (elems * esz + 255) & ~(size_t)255;
    return p;
  };
  const i64 NQ = (i64)Bsz * T1 * FEAT;       // 2,097,152
  const i64 NW = (i64)FEAT * FEAT;           // 262,144
  unsigned short* qbf  = (unsigned short*)alloc(NQ, 2);
  unsigned short* kbf  = (unsigned short*)alloc(NQ, 2);
  unsigned short* vbf  = (unsigned short*)alloc(NQ, 2);
  unsigned short* pebf = (unsigned short*)alloc((i64)NPAD * FEAT, 2);
  unsigned short* wq   = (unsigned short*)alloc(NW, 2);
  unsigned short* wk   = (unsigned short*)alloc(NW, 2);
  unsigned short* wv   = (unsigned short*)alloc(NW, 2);
  unsigned short* wo   = (unsigned short*)alloc(NW, 2);
  unsigned short* wpos = (unsigned short*)alloc(NW, 2);
  unsigned short* qu   = (unsigned short*)alloc((i64)BH * T1 * DK, 2);
  unsigned short* qv   = (unsigned short*)alloc((i64)BH * T1 * DK, 2);
  unsigned short* kb   = (unsigned short*)alloc((i64)BH * T2 * DK, 2);
  unsigned short* vT   = (unsigned short*)alloc((i64)BH * DK * T2, 2);
  unsigned short* pb   = (unsigned short*)alloc((i64)H * NPAD * DK, 2);
  unsigned short* xbf  = (unsigned short*)alloc(NQ, 2);

  // conversions
  k_cvt<<<(int)((NQ + 255) / 256), 256, 0, stream>>>(query, qbf, (int)NQ);
  k_cvt<<<(int)((NQ + 255) / 256), 256, 0, stream>>>(key,   kbf, (int)NQ);
  k_cvt<<<(int)((NQ + 255) / 256), 256, 0, stream>>>(value, vbf, (int)NQ);
  k_cvt<<<(int)((NW + 255) / 256), 256, 0, stream>>>(Wq,   wq,   (int)NW);
  k_cvt<<<(int)((NW + 255) / 256), 256, 0, stream>>>(Wk,   wk,   (int)NW);
  k_cvt<<<(int)((NW + 255) / 256), 256, 0, stream>>>(Wv,   wv,   (int)NW);
  k_cvt<<<(int)((NW + 255) / 256), 256, 0, stream>>>(Wo,   wo,   (int)NW);
  k_cvt<<<(int)((NW + 255) / 256), 256, 0, stream>>>(Wpos, wpos, (int)NW);
  k_cvt_pad<<<(NPAD * FEAT + 255) / 256, 256, 0, stream>>>(pos_emb, pebf,
                                                           NPOS * FEAT, NPAD * FEAT);
  // cache -> new_cache head + bf16 k / v^T
  {
    i64 n = (i64)BH * CACHE_T * 2 * DK;
    k_cache<<<(int)((n + 255) / 256), 256, 0, stream>>>(cache, new_cache, kb, vT);
  }
  // projections (each wave = 16x64 strip)
  k_qproj<<<512, 128, 0, stream>>>(qbf, wq, bq, pbu, pbv, qu, qv);
  k_kvproj<<<512, 128, 0, stream>>>(kbf, wk, bk, new_cache, kb, vT, 0);
  k_kvproj<<<512, 128, 0, stream>>>(vbf, wv, bv, new_cache, kb, vT, 1);
  k_pproj<<<256, 128, 0, stream>>>(pebf, wpos, pb);
  // fused attention: 32 i-tiles per (b,h), 4 waves/block
  k_attn<<<dim3(8, 1, 64), 128, 0, stream>>>(qu, qv, kb, pb, vT, mask, xbf);
  // out = x @ Wo^T + bo
  k_out<<<512, 128, 0, stream>>>(xbf, wo, bo, out);
}
